// SelfAttention_23416161698171
// MI455X (gfx1250) — matrix operand order repaired and checked by reading, not compiled, not run on hardware
//
#include <hip/hip_runtime.h>
#include <hip/hip_bf16.h>

// ---------------------------------------------------------------------------
// Shapes (fixed by the reference): B=1, R=64, C=384, D=256, H=8, hd=32, P=128
// ---------------------------------------------------------------------------
#define RR   64
#define CC   384
#define DD   256
#define HH   8
#define HD   32
#define PP   128
#define MM   (RR * CC)          // 24576 token rows
#define NTOT (4 * DD)           // Q|K|V|G concatenated along N = 1024
#define SCALING 0.17677669529663687f

typedef __attribute__((ext_vector_type(16))) _Float16 v16h;
typedef __attribute__((ext_vector_type(8)))  _Float16 v8h;
typedef __attribute__((ext_vector_type(4)))  _Float16 v4h;
typedef __attribute__((ext_vector_type(8)))  float    v8f;

#define WMMA(a, b, c) __builtin_amdgcn_wmma_f32_16x16x32_f16( \
    false, (a), false, (b), (short)0, (c), false, false)

// A fragment: 16x32 (MxK) f16, row-major source with leading dim `ld` (halves).
// lane<16 : row=lane,    K[0..15]  contiguous
// lane>=16: row=lane-16, K[16..31] contiguous
static __device__ __forceinline__ v16h load_a_frag(const _Float16* base, int ld) {
  const int lane = threadIdx.x & 31;
  const int row  = lane & 15;
  const int g    = lane >> 4;
  const _Float16* p = base + (size_t)row * ld;
  v8h lo = *(const v8h*)(p + g * 16);
  v8h hi = *(const v8h*)(p + g * 16 + 8);
  v16h a;
#pragma unroll
  for (int i = 0; i < 8; ++i) { a[i] = lo[i]; a[8 + i] = hi[i]; }
  return a;
}

// B fragment: 32x16 (KxN) f16 from a K-contiguous ("transposed") buffer
// bt[n][k] with leading dim `ld` (halves): lane holds column n=lane&15,
// K chunk [ (lane>>4)*16 .. +15 ] contiguous -> single 32-byte load.
static __device__ __forceinline__ v16h load_b_frag(const _Float16* bt, int ld) {
  const int lane = threadIdx.x & 31;
  const int n = lane & 15;
  const int g = lane >> 4;
  return *(const v16h*)(bt + (size_t)n * ld + g * 16);
}

// ---------------------------------------------------------------------------
// Kernel 1: x (f32) -> Xh (f16)
// ---------------------------------------------------------------------------
__global__ __launch_bounds__(256) void conv_x_kernel(
    const float* __restrict__ x, _Float16* __restrict__ xh) {
  int t = blockIdx.x * 256 + threadIdx.x;          // one float4 per thread
  float4 v = ((const float4*)x)[t];
  v4h o; o[0] = (_Float16)v.x; o[1] = (_Float16)v.y;
  o[2] = (_Float16)v.z; o[3] = (_Float16)v.w;
  ((v4h*)xh)[t] = o;
}

// ---------------------------------------------------------------------------
// Kernel 2: pack Wq|Wk|Wv|Wg (f32 [K=256][N=256]) -> Wt f16 [gn=0..1023][k]
// Wt[(mat*256+n)*256 + k] = W_mat[k*256 + n]  (transposed, K-contiguous)
// ---------------------------------------------------------------------------
__global__ __launch_bounds__(256) void conv_w_kernel(
    const float* __restrict__ Wq, const float* __restrict__ Wk,
    const float* __restrict__ Wv, const float* __restrict__ Wg,
    _Float16* __restrict__ Wt) {
  int t = blockIdx.x * 256 + threadIdx.x;          // 0 .. 262143
  int mat = t >> 16;
  int rem = t & 65535;
  int n = rem >> 8, k = rem & 255;
  const float* W = (mat == 0) ? Wq : (mat == 1) ? Wk : (mat == 2) ? Wv : Wg;
  Wt[t] = (_Float16)W[k * 256 + n];
}

// ---------------------------------------------------------------------------
// Kernel 3: fused projection GEMM  Y[M=24576][N=1024] = Xh @ [Wq|Wk|Wv|Wg]
// Block = 8 waves, 64x32 tile (waves in 4x2). One v_wmma per 32-K step.
// Epilogue scatters per matrix:
//   Q: *SCALING -> Qb [r*8+h][c][32] f16
//   K:          -> Kb [r*8+h][c][32] f16
//   V:          -> Vt [r*8+h][32][c] f16 (d-major for contiguous B-frags)
//   G: sigmoid(+bg) -> Gb [m][256] f16
// ---------------------------------------------------------------------------
__global__ __launch_bounds__(256) void gemm_qkvg_kernel(
    const _Float16* __restrict__ Xh, const _Float16* __restrict__ Wt,
    const float* __restrict__ bg,
    _Float16* __restrict__ Qb, _Float16* __restrict__ Kb,
    _Float16* __restrict__ Vt, _Float16* __restrict__ Gb) {
  const int wid  = threadIdx.x >> 5;
  const int lane = threadIdx.x & 31;
  const int wy = wid & 3, wx = wid >> 2;
  const int m0  = blockIdx.x * 64 + wy * 16;
  const int gn0 = blockIdx.y * 32 + wx * 16;

  const _Float16* a_base = Xh + (size_t)m0 * DD;
  const _Float16* b_base = Wt + (size_t)gn0 * DD;

  v8f c = {};
#pragma unroll
  for (int k0 = 0; k0 < DD; k0 += 32) {
    v16h a = load_a_frag(a_base + k0, DD);
    v16h b = load_b_frag(b_base + k0, DD);
    c = WMMA(a, b, c);
  }

  const int nl  = lane & 15;
  const int g   = lane >> 4;
  const int gn  = gn0 + nl;
  const int mat = gn >> 8;       // uniform per wave
  const int n   = gn & 255;
  const int h   = n >> 5, d = n & 31;

#pragma unroll
  for (int v = 0; v < 8; ++v) {
    const int gm = m0 + g * 8 + v;
    const int r = gm / CC, cc = gm % CC;
    float val = c[v];
    if (mat == 0) {
      Qb[((size_t)(r * HH + h) * CC + cc) * HD + d] = (_Float16)(val * SCALING);
    } else if (mat == 1) {
      Kb[((size_t)(r * HH + h) * CC + cc) * HD + d] = (_Float16)val;
    } else if (mat == 2) {
      Vt[((size_t)(r * HH + h) * HD + d) * CC + cc] = (_Float16)val;
    } else {
      float s = 1.0f / (1.0f + __expf(-(val + bg[n])));
      Gb[(size_t)gm * DD + n] = (_Float16)s;
    }
  }
}

// ---------------------------------------------------------------------------
// Kernel 4: pair bias = LayerNorm(pair_feat) @ Wb, mask folded in.
// One wave per (i,j) row of 128; output biasE[h][i][j] f32 (or -1e30 masked).
// ---------------------------------------------------------------------------
__global__ __launch_bounds__(256) void pair_bias_kernel(
    const float* __restrict__ pair, const int* __restrict__ mask,
    const float* __restrict__ ln_g, const float* __restrict__ ln_b,
    const float* __restrict__ Wb, float* __restrict__ biasE) {
  const int wid  = threadIdx.x >> 5;
  const int lane = threadIdx.x & 31;
  const long row = (long)blockIdx.x * 8 + wid;      // 0 .. 147455
  const int i = (int)(row / CC), j = (int)(row % CC);

  float4 xv = ((const float4*)(pair + row * PP))[lane];
  float x0 = xv.x, x1 = xv.y, x2 = xv.z, x3 = xv.w;

  float s  = x0 + x1 + x2 + x3;
  float ss = x0 * x0 + x1 * x1 + x2 * x2 + x3 * x3;
#pragma unroll
  for (int mk = 1; mk < 32; mk <<= 1) {
    s  += __shfl_xor(s, mk);
    ss += __shfl_xor(ss, mk);
  }
  const float mu  = s * (1.0f / PP);
  const float var = ss * (1.0f / PP) - mu * mu;
  const float inv = rsqrtf(var + 1e-5f);

  const int p0 = lane * 4;
  float n0 = (x0 - mu) * inv * ln_g[p0 + 0] + ln_b[p0 + 0];
  float n1 = (x1 - mu) * inv * ln_g[p0 + 1] + ln_b[p0 + 1];
  float n2 = (x2 - mu) * inv * ln_g[p0 + 2] + ln_b[p0 + 2];
  float n3 = (x3 - mu) * inv * ln_g[p0 + 3] + ln_b[p0 + 3];

  float acc[HH];
#pragma unroll
  for (int h = 0; h < HH; ++h) {
    acc[h] = n0 * Wb[(p0 + 0) * HH + h] + n1 * Wb[(p0 + 1) * HH + h] +
             n2 * Wb[(p0 + 2) * HH + h] + n3 * Wb[(p0 + 3) * HH + h];
#pragma unroll
    for (int mk = 1; mk < 32; mk <<= 1) acc[h] += __shfl_xor(acc[h], mk);
  }

  const bool ok = (mask[i] != 0) && (mask[j] != 0);
  if (lane == 0) {
#pragma unroll
    for (int h = 0; h < HH; ++h)
      biasE[((size_t)h * CC + i) * CC + j] = ok ? acc[h] : -1e30f;
  }
}

// ---------------------------------------------------------------------------
// Kernel 5: flash attention per (r,h). Block = 8 waves x 16 query rows = 128
// rows; 3 blocks cover C=384. Per 32-col tile: 2 WMMA (S), online softmax,
// LDS re-fragment of P, 2 WMMA (O += P@V). Epilogue: /l, *gate, f32 store.
// ---------------------------------------------------------------------------
__global__ __launch_bounds__(256) void attn_kernel(
    const _Float16* __restrict__ Qb, const _Float16* __restrict__ Kb,
    const _Float16* __restrict__ Vt, const _Float16* __restrict__ Gb,
    const float* __restrict__ biasE, float* __restrict__ out) {
  __shared__ __align__(32) _Float16 pshare[8][16 * 32];   // 8 KB, 1 KB/wave

  const int wid  = threadIdx.x >> 5;
  const int lane = threadIdx.x & 31;
  const int rh     = blockIdx.x / 3;       // r*8 + h
  const int ichunk = blockIdx.x % 3;
  const int r = rh >> 3, h = rh & 7;
  const int i0 = ichunk * 128 + wid * 16;

  const _Float16* Qp = Qb + ((size_t)rh * CC + i0) * HD;
  const _Float16* Kp = Kb + (size_t)rh * CC * HD;
  const _Float16* Vp = Vt + (size_t)rh * HD * CC;
  const float*    bp = biasE + (size_t)h * CC * CC;

  const v16h aq = load_a_frag(Qp, HD);
  const int nl = lane & 15, g = lane >> 4;

  float m[8], l[8];
  v8f o0 = {}, o1 = {};
#pragma unroll
  for (int v = 0; v < 8; ++v) { m[v] = -3.0e38f; l[v] = 0.0f; }

  for (int j0 = 0; j0 < CC; j0 += 32) {
    v16h bk0 = load_b_frag(Kp + (size_t)j0 * HD, HD);
    v16h bk1 = load_b_frag(Kp + (size_t)(j0 + 16) * HD, HD);
    v8f s0 = {}, s1 = {};
    s0 = WMMA(aq, bk0, s0);
    s1 = WMMA(aq, bk1, s1);

    const float* brow = bp + (size_t)(i0 + g * 8) * CC + j0 + nl;
    _Float16* sh = pshare[wid];
#pragma unroll
    for (int v = 0; v < 8; ++v) {
      float t0 = s0[v] + brow[(size_t)v * CC];
      float t1 = s1[v] + brow[(size_t)v * CC + 16];
      // row max across the 16 lanes sharing this row set
      float mx = fmaxf(t0, t1);
#pragma unroll
      for (int mk = 1; mk < 16; mk <<= 1) mx = fmaxf(mx, __shfl_xor(mx, mk));
      const float nm    = fmaxf(m[v], mx);
      const float alpha = __expf(m[v] - nm);
      m[v] = nm;
      const float q0 = __expf(t0 - nm);
      const float q1 = __expf(t1 - nm);
      float rs = q0 + q1;
#pragma unroll
      for (int mk = 1; mk < 16; mk <<= 1) rs += __shfl_xor(rs, mk);
      l[v]  = l[v] * alpha + rs;
      o0[v] *= alpha;
      o1[v] *= alpha;
      _Float16* prow = sh + (g * 8 + v) * 32;
      prow[nl]      = (_Float16)q0;
      prow[16 + nl] = (_Float16)q1;
    }
    // re-fragment P (16x32) as an A matrix via this wave's private LDS slab
    v16h ap  = load_a_frag(sh, 32);
    v16h bv0 = load_b_frag(Vp + j0, CC);             // d = 0..15
    v16h bv1 = load_b_frag(Vp + (size_t)16 * CC + j0, CC); // d = 16..31
    o0 = WMMA(ap, bv0, o0);
    o1 = WMMA(ap, bv1, o1);
  }

#pragma unroll
  for (int v = 0; v < 8; ++v) {
    const float invl = 1.0f / l[v];
    const int ci = i0 + g * 8 + v;
    const size_t base = ((size_t)r * CC + ci) * DD + h * HD;
    const float g0 = (float)Gb[base + nl];
    const float g1 = (float)Gb[base + 16 + nl];
    out[base + nl]      = g0 * o0[v] * invl;
    out[base + 16 + nl] = g1 * o1[v] * invl;
  }
}

// ---------------------------------------------------------------------------
// Host launcher
// ---------------------------------------------------------------------------
extern "C" void kernel_launch(void* const* d_in, const int* in_sizes, int n_in,
                              void* d_out, int out_size, void* d_ws, size_t ws_size,
                              hipStream_t stream) {
  (void)in_sizes; (void)n_in; (void)out_size; (void)ws_size;
  const float* x    = (const float*)d_in[0];
  const float* pair = (const float*)d_in[1];
  const int*   msk  = (const int*)d_in[2];
  const float* Wq   = (const float*)d_in[3];
  const float* Wk   = (const float*)d_in[4];
  const float* Wv   = (const float*)d_in[5];
  const float* lg   = (const float*)d_in[6];
  const float* lb   = (const float*)d_in[7];
  const float* Wb   = (const float*)d_in[8];
  const float* Wg   = (const float*)d_in[9];
  const float* bg   = (const float*)d_in[10];
  float* out = (float*)d_out;

  // workspace carve-up (all offsets 256B-aligned by construction)
  char* ws = (char*)d_ws;
  size_t off = 0;
  _Float16* Xh = (_Float16*)(ws + off); off += (size_t)MM * DD * 2;       // 12.6 MB
  _Float16* Wt = (_Float16*)(ws + off); off += (size_t)NTOT * DD * 2;     // 0.5 MB
  _Float16* Qb = (_Float16*)(ws + off); off += (size_t)RR*HH*CC*HD * 2;   // 12.6 MB
  _Float16* Kb = (_Float16*)(ws + off); off += (size_t)RR*HH*CC*HD * 2;   // 12.6 MB
  _Float16* Vt = (_Float16*)(ws + off); off += (size_t)RR*HH*HD*CC * 2;   // 12.6 MB
  _Float16* Gb = (_Float16*)(ws + off); off += (size_t)MM * DD * 2;       // 12.6 MB
  float*  biasE = (float*)(ws + off);   off += (size_t)HH * CC * CC * 4;  // 4.7 MB

  conv_x_kernel<<<(MM * DD) / (256 * 4), 256, 0, stream>>>(x, Xh);
  conv_w_kernel<<<(NTOT * DD) / 256, 256, 0, stream>>>(Wq, Wk, Wv, Wg, Wt);
  gemm_qkvg_kernel<<<dim3(MM / 64, NTOT / 32), 256, 0, stream>>>(
      Xh, Wt, bg, Qb, Kb, Vt, Gb);
  pair_bias_kernel<<<(CC * CC) / 8, 256, 0, stream>>>(pair, msk, lg, lb, Wb, biasE);
  attn_kernel<<<RR * HH * 3, 256, 0, stream>>>(Qb, Kb, Vt, Gb, biasE, out);
}
